// Engram_22823456211649
// MI455X (gfx1250) — matrix operand order repaired and checked by reading, not compiled, not run on hardware
//
#include <hip/hip_runtime.h>

// ---------------- problem constants ----------------
#define LSEQ 4096
#define BATCH 2
#define DDIM 2048
#define HHEADS 12
#define DH 128
#define EDIM 1536          // H*Dh
#define NVOCAB 100000
#define MROWS (LSEQ*BATCH) // 8192 GEMM rows

// ---------------- GEMM tiling ----------------
#define BM 32                      // rows per workgroup
#define ROWB 3120                  // LDS bytes per padded A row: 3072 + 3*16
#define SLAB (BM*ROWB)             // 99840 B
#define SMEM_BYTES (SLAB + 128*4)  // + stats (skk/skh/shh/gate, 32 each)

typedef __bf16 v16bf __attribute__((ext_vector_type(16)));
typedef __bf16 v8bf  __attribute__((ext_vector_type(8)));
typedef __bf16 v4bf  __attribute__((ext_vector_type(4)));
typedef float  v8f   __attribute__((ext_vector_type(8)));
typedef unsigned int v4u __attribute__((ext_vector_type(4)));
typedef int v8i __attribute__((ext_vector_type(8)));
typedef int v4i __attribute__((ext_vector_type(4)));

// ---------------- kernel 1: fp32 -> bf16 weight conversion ----------------
__global__ void engram_cvt_bf16(const float* __restrict__ src,
                                __bf16* __restrict__ dst, int n) {
  int i = (blockIdx.x * blockDim.x + threadIdx.x) * 4;
  int stride = gridDim.x * blockDim.x * 4;
  for (; i < n; i += stride) {
    float4 v = *(const float4*)(src + i);
    v4bf o = {(__bf16)v.x, (__bf16)v.y, (__bf16)v.z, (__bf16)v.w};
    *(v4bf*)(dst + i) = o;
  }
}

// ---------------- kernel 2: hash gather -> bf16 A matrix ----------------
// one wave per (l,b,h) row; 128 floats -> 128 bf16
__global__ void engram_gather(const float* __restrict__ tables,
                              const int* __restrict__ ids,
                              __bf16* __restrict__ dst) {
  int r = blockIdx.x * 8 + (threadIdx.x >> 5);   // (l*B+b)*H + h
  int lane = threadIdx.x & 31;
  int h = r % HHEADS;
  int row = r / HHEADS;
  int id = ids[r];
  const float4* src =
      (const float4*)(tables + ((size_t)h * NVOCAB + (size_t)id) * DH) + lane;
  float4 v = *src;
  v4bf o = {(__bf16)v.x, (__bf16)v.y, (__bf16)v.z, (__bf16)v.w};
  *((v4bf*)(dst + (size_t)row * EDIM + h * DH) + lane) = o;
}

// ---------------- kernel 3: fused key/value GEMM + gate ----------------
__global__ __launch_bounds__(256, 1) void engram_gemm_kernel(
    const float* __restrict__ hidden,      // [8192,2048]
    const __bf16* __restrict__ embbf,      // [8192,1536]
    const __bf16* __restrict__ kwbf,       // [2048,1536]
    const __bf16* __restrict__ vwbf,       // [2048,1536]
    const float* __restrict__ key_b,       // [2048]
    const float* __restrict__ value_b,     // [2048]
    const float* __restrict__ n1w,         // [2048]
    const float* __restrict__ n2w,         // [2048]
    float* __restrict__ value_out)         // [8192,2048]
{
  extern __shared__ __align__(16) unsigned char smem[];
  float* stats = (float*)(smem + SLAB);
  float* skk = stats;         // sum k^2 per row
  float* skh = stats + 32;    // sum k*h*w1*w2 per row
  float* shh = stats + 64;    // sum h^2 per row
  float* gateA = stats + 96;  // per-row gate

  const int tid  = threadIdx.x;
  const int wave = tid >> 5;
  const int lane = tid & 31;
  const int lh   = lane >> 4;   // lane half (K-split of fragments)
  const int ln   = lane & 15;
  const int rb   = wave & 1;    // row block (16 rows each)
  const int cg   = wave >> 1;   // column group (0..3)
  const int m0   = blockIdx.x * BM;

  if (tid < 128) stats[tid] = 0.f;

  // ---- A-slab: global -> LDS via Tensor Data Mover (padded rows) ----
#if __has_builtin(__builtin_amdgcn_tensor_load_to_lds) && __has_builtin(__builtin_amdgcn_s_wait_tensorcnt)
  if (wave == 0) {
    unsigned lds_base = (unsigned)(unsigned long long)(uintptr_t)(void*)smem;
    unsigned long long ga =
        (unsigned long long)(uintptr_t)(embbf + (size_t)m0 * EDIM);
    v4u g0 = {1u,                                   // count=1, user desc
              lds_base,                             // lds_addr
              (unsigned)(ga & 0xffffffffu),         // global_addr lo
              (unsigned)((ga >> 32) & 0x1ffffffu) | (2u << 30)}; // hi | type=2
    v8i g1;
    g1[0] = (1 << 16)        // data_size = 2 bytes (bf16)
          | (1 << 20)        // pad_enable
          | (7 << 22)        // pad_interval: 256 dwords = 1024 B
          | (3 << 25);       // pad_amount: 4 dwords = 16 B
    g1[1] = (int)((unsigned)EDIM << 16);       // tensor_dim0 lo16
    g1[2] = (int)((unsigned)BM << 16);         // dim0 hi | tensor_dim1 lo16
    g1[3] = (int)((unsigned)EDIM << 16);       // dim1 hi | tile_dim0
    g1[4] = BM;                                // tile_dim1 | tile_dim2=0
    g1[5] = EDIM;                              // tensor_dim0_stride lo32
    g1[6] = 0;
    g1[7] = 0;
    v4i z4 = {0, 0, 0, 0};
    v8i z8 = {0, 0, 0, 0, 0, 0, 0, 0};
    __builtin_amdgcn_tensor_load_to_lds(g0, g1, z4, z4, z8, 0);
  }
  __syncthreads();   // stats zero visible before atomics below
#else
  // fallback: cooperative copy honoring the padded LDS layout
  {
    const __bf16* src = embbf + (size_t)m0 * EDIM;
    for (int c = tid; c < BM * (EDIM / 8); c += 256) {
      int r = c / (EDIM / 8);
      int ce = (c % (EDIM / 8)) * 8;
      unsigned byt = 2u * (unsigned)ce;
      byt += (byt >> 10) << 4;
      *(v8bf*)(smem + r * ROWB + byt) = *(const v8bf*)(src + (size_t)r * EDIM + ce);
    }
  }
  __syncthreads();
#endif

  // ---- per-row sum(h^2) while the TDM is in flight ----
  {
    int r = tid >> 3, sub = tid & 7;
    const float* hp = hidden + (size_t)(m0 + r) * DDIM + sub * 256;
    float s = 0.f;
    for (int i = 0; i < 256; i += 4) {
      float4 hv = *(const float4*)(hp + i);
      s += hv.x * hv.x + hv.y * hv.y + hv.z * hv.z + hv.w * hv.w;
    }
    atomicAdd(&shh[r], s);
  }

#if __has_builtin(__builtin_amdgcn_tensor_load_to_lds) && __has_builtin(__builtin_amdgcn_s_wait_tensorcnt)
  if (wave == 0) __builtin_amdgcn_s_wait_tensorcnt(0);
#endif
  __syncthreads();   // A slab ready for everyone

  // A fragment: lane ln supplies row rb*16+ln; lh selects K half 16*lh..16*lh+15
  const unsigned aRowOff = (unsigned)((rb * 16 + ln) * ROWB);

  for (int pass = 0; pass < 2; ++pass) {
    const __bf16* W = pass ? vwbf : kwbf;
    const float* bias = pass ? value_b : key_b;

    for (int ni = 0; ni < 16; ++ni) {
      const int nb0 = ni * 128 + cg * 32;
      const __bf16* bp0 = W + (size_t)(nb0 + ln) * EDIM;        // tile0 column
      const __bf16* bp1 = bp0 + (size_t)16 * EDIM;              // tile1 column
      v8f acc0 = {};
      v8f acc1 = {};

#pragma unroll 4
      for (int kk = 0; kk < EDIM; kk += 32) {
        // A k order used here (contiguous, not AMD table order): lane<16 -> K{0..15}, lane>=16 -> K{16..31}
        const int e_lo = kk + lh * 16;
        const int e_hi = kk + 8 + lh * 16;
        unsigned bo_lo = 2u * (unsigned)e_lo; bo_lo += (bo_lo >> 10) << 4;
        unsigned bo_hi = 2u * (unsigned)e_hi; bo_hi += (bo_hi >> 10) << 4;
        v8bf alo = *(const v8bf*)(smem + aRowOff + bo_lo);
        v8bf ahi = *(const v8bf*)(smem + aRowOff + bo_hi);
        // B k order used here (contiguous, same as A): lanes 0-15 hold K=kk..kk+15, lanes 16-31 K=kk+16..kk+31
        v16bf b0 = *(const v16bf*)(bp0 + kk + 16 * lh);
        v16bf b1 = *(const v16bf*)(bp1 + kk + 16 * lh);
        v16bf a;
#pragma unroll
        for (int t = 0; t < 8; ++t) { a[t] = alo[t]; a[t + 8] = ahi[t]; }
        acc0 = __builtin_amdgcn_wmma_f32_16x16x32_bf16(
            false, a, false, b0, (short)0, acc0, false, false);
        acc1 = __builtin_amdgcn_wmma_f32_16x16x32_bf16(
            false, a, false, b1, (short)0, acc1, false, false);
      }

      if (pass == 0) {
        // accumulate per-row sum(k^2) and sum(k*h*w1*w2)
        auto do_stats = [&](const v8f& A, int colbase) {
          int col = colbase + ln;
          float w12 = n1w[col] * n2w[col];
          float bcol = bias[col];
          const float* hcol = hidden + col;
#pragma unroll
          for (int j = 0; j < 8; ++j) {
            int rl = rb * 16 + j + 8 * lh;     // C layout: lanes>=16 hold M=j+8
            float kv = A[j] + bcol;
            float h = hcol[(size_t)(m0 + rl) * DDIM];
            float s1 = kv * kv;
            float s2 = kv * h * w12;
            s1 += __shfl_xor(s1, 1, 32);  s2 += __shfl_xor(s2, 1, 32);
            s1 += __shfl_xor(s1, 2, 32);  s2 += __shfl_xor(s2, 2, 32);
            s1 += __shfl_xor(s1, 4, 32);  s2 += __shfl_xor(s2, 4, 32);
            s1 += __shfl_xor(s1, 8, 32);  s2 += __shfl_xor(s2, 8, 32);
            if (ln == 0) {
              atomicAdd(&skk[rl], s1);
              atomicAdd(&skh[rl], s2);
            }
          }
        };
        do_stats(acc0, nb0);
        do_stats(acc1, nb0 + 16);
      } else {
        // gated value write-out
        auto do_store = [&](const v8f& A, int colbase) {
          int col = colbase + ln;
          float bcol = bias[col];
#pragma unroll
          for (int j = 0; j < 8; ++j) {
            int rl = rb * 16 + j + 8 * lh;
            float v = gateA[rl] * (A[j] + bcol);
            value_out[(size_t)(m0 + rl) * DDIM + col] = v;
          }
        };
        do_store(acc0, nb0);
        do_store(acc1, nb0 + 16);
      }
    }

    if (pass == 0) {
      __syncthreads();
      if (tid < 32) {
        float rsk = rsqrtf(skk[tid] * (1.0f / DDIM) + 1e-6f);
        float rsq = rsqrtf(shh[tid] * (1.0f / DDIM) + 1e-6f);
        float g = rsk * rsq * skh[tid] * 0.022097086912079608f; // 1/sqrt(2048)
        float ag = fmaxf(fabsf(g), 1e-6f);
        g = copysignf(sqrtf(ag), g);
        gateA[tid] = 1.f / (1.f + __expf(-g));
      }
      __syncthreads();
    }
  }
}

// ---------------- kernel 4: dilated causal depthwise conv + residual ----------------
__global__ void engram_conv(const float* __restrict__ val,
                            const float* __restrict__ convw,  // [D,4]
                            const float* __restrict__ convb,  // [D]
                            float* __restrict__ out) {
  size_t idx = (size_t)blockIdx.x * blockDim.x + threadIdx.x;
  int d = (int)(idx & (DDIM - 1));
  int row = (int)(idx >> 11);
  int l = row >> 1;  // B == 2
  float x = val[idx];
  float y = convb[d] + convw[d * 4 + 3] * x;                    // k=3 -> lag 0
  if (l >= 4)  y += convw[d * 4 + 2] * val[idx - (size_t)8  * DDIM];
  if (l >= 8)  y += convw[d * 4 + 1] * val[idx - (size_t)16 * DDIM];
  if (l >= 12) y += convw[d * 4 + 0] * val[idx - (size_t)24 * DDIM];
  out[idx] = x + y;
}

// ---------------- launch ----------------
extern "C" void kernel_launch(void* const* d_in, const int* in_sizes, int n_in,
                              void* d_out, int out_size, void* d_ws, size_t ws_size,
                              hipStream_t stream) {
  (void)in_sizes; (void)n_in; (void)out_size; (void)ws_size;
  const float* hidden = (const float*)d_in[0];
  const int*   ids    = (const int*)d_in[1];
  const float* tables = (const float*)d_in[2];
  const float* key_w  = (const float*)d_in[3];
  const float* key_b  = (const float*)d_in[4];
  const float* n1w    = (const float*)d_in[5];
  const float* n2w    = (const float*)d_in[6];
  const float* val_w  = (const float*)d_in[7];
  const float* val_b  = (const float*)d_in[8];
  const float* conv_w = (const float*)d_in[9];
  const float* conv_b = (const float*)d_in[10];
  float* out = (float*)d_out;

  // workspace layout (bytes):
  //   [0, 25165824)            emb bf16   [8192,1536]
  //   [25165824, 31457280)     key_w bf16 [2048,1536]
  //   [31457280, 37748736)     value_w bf16
  //   [37748736, 104857600)    value f32  [8192,2048]
  char* ws = (char*)d_ws;
  __bf16* embbf = (__bf16*)ws;
  __bf16* kwbf  = (__bf16*)(ws + 25165824);
  __bf16* vwbf  = (__bf16*)(ws + 31457280);
  float*  valf  = (float*)(ws + 37748736);

  (void)hipFuncSetAttribute((const void*)engram_gemm_kernel,
                            hipFuncAttributeMaxDynamicSharedMemorySize,
                            SMEM_BYTES);

  engram_cvt_bf16<<<3072, 256, 0, stream>>>(key_w, kwbf, DDIM * EDIM);
  engram_cvt_bf16<<<3072, 256, 0, stream>>>(val_w, vwbf, DDIM * EDIM);
  engram_gather<<<(MROWS * HHEADS) / 8, 256, 0, stream>>>(tables, ids, embbf);
  engram_gemm_kernel<<<MROWS / BM, 256, SMEM_BYTES, stream>>>(
      hidden, embbf, kwbf, vwbf, key_b, val_b, n1w, n2w, valf);
  engram_conv<<<(MROWS * DDIM) / 256, 256, 0, stream>>>(valf, conv_w, conv_b, out);
}
